// SG_Attention_81690277970121
// MI455X (gfx1250) — matrix operand order repaired and checked by reading, not compiled, not run on hardware
//
#include <hip/hip_runtime.h>
#include <hip/hip_bf16.h>

typedef __attribute__((ext_vector_type(16))) _Float16 v16h;
typedef __attribute__((ext_vector_type(8)))  _Float16 v8h;
typedef __attribute__((ext_vector_type(8)))  float    v8f;

// Problem constants (B, H0, W0, D0 = 2, 28, 28, 28; C=96, NH=6, SR=4, WS=7)
#define NB    2
#define CC    96
#define NH2   3          // heads per branch
#define HD    16         // head dim -> one WMMA tile
#define NN    21952      // 28^3
#define MR    43904      // B * N rows
#define NK    343        // 7^3 (SR-reduced tokens == window size)
#define KT22  22         // ceil(343/16)
#define QT1   1372       // 21952/16 query tiles per (b,h) in branch 1

#define CAT16(a, b) __builtin_shufflevector(a, b, 0,1,2,3,4,5,6,7,8,9,10,11,12,13,14,15)

// Convert 8 floats (two float4) into v16h elements [base, base+8)
__device__ __forceinline__ void cvt8(const float4 f0, const float4 f1,
                                     v16h& v, const int base) {
  v[base + 0] = (_Float16)f0.x; v[base + 1] = (_Float16)f0.y;
  v[base + 2] = (_Float16)f0.z; v[base + 3] = (_Float16)f0.w;
  v[base + 4] = (_Float16)f1.x; v[base + 5] = (_Float16)f1.y;
  v[base + 6] = (_Float16)f1.z; v[base + 7] = (_Float16)f1.w;
}

// f32 -> f16 conversion pass (amortized once per buffer)
__global__ __launch_bounds__(256) void cvt_kernel(const float* __restrict__ s,
                                                  _Float16* __restrict__ d, int n) {
  const int idx = blockIdx.x * 256 + threadIdx.x;
  if (idx < n) d[idx] = (_Float16)s[idx];
}

// ---------------------------------------------------------------------------
// WMMA GEMM, f16 operands: C[M,Nc] = A[M,K] * W[Nc,K]^T + bias
// One wave computes a 32x16 output block (2 M-tiles share one B operand).
// REQUIRES K % 32 == 0.  Output f16 (Ch) or f32 (Cf) -- exactly one non-null.
// ---------------------------------------------------------------------------
__global__ __launch_bounds__(128) void wmma_gemm_f16(
    const _Float16* __restrict__ A, const _Float16* __restrict__ W,
    const float* __restrict__ bias, float* __restrict__ Cf,
    _Float16* __restrict__ Ch, int M, int Nc, int K, int lda) {
  const int lane = threadIdx.x & 31;
  const int half = lane >> 4;
  const int l    = lane & 15;
  const int mt  = (M + 15) >> 4;
  const int mt2 = (mt + 1) >> 1;              // 32-row blocks
  const int nt  = Nc >> 4;
  const int tile = blockIdx.x * 4 + (threadIdx.x >> 5);
  if (tile >= mt2 * nt) return;               // wave-uniform, no barriers here
  const int tm2 = tile / nt;
  const int tn  = tile % nt;

  int row0 = tm2 * 32 + l;      if (row0 >= M) row0 = M - 1;
  int row1 = tm2 * 32 + 16 + l; if (row1 >= M) row1 = M - 1;
  const int bcol = tn * 16 + l;
  const _Float16* Ap0 = A + (size_t)row0 * lda;
  const _Float16* Ap1 = A + (size_t)row1 * lda;
  const _Float16* Wp  = W + (size_t)bcol * K;

  v8f acc0 = {}, acc1 = {};
  for (int k0 = 0; k0 < K; k0 += 32) {
    if (k0 + 32 < K) __builtin_prefetch(Wp + k0 + 32, 0, 3);
    const v8h a00 = *(const v8h*)(Ap0 + k0 + (half << 3));
    const v8h a01 = *(const v8h*)(Ap0 + k0 + (half << 3) + 16);
    const v8h a10 = *(const v8h*)(Ap1 + k0 + (half << 3));
    const v8h a11 = *(const v8h*)(Ap1 + k0 + (half << 3) + 16);
    const v8h b0  = *(const v8h*)(Wp  + k0 + (half << 3));
    const v8h b1  = *(const v8h*)(Wp  + k0 + (half << 3) + 16);
    const v16h va0 = CAT16(a00, a01);
    const v16h va1 = CAT16(a10, a11);
    const v16h vb  = CAT16(b0, b1);
    acc0 = __builtin_amdgcn_wmma_f32_16x16x32_f16(false, va0, false, vb,
                                                  (short)0, acc0, false, false);
    acc1 = __builtin_amdgcn_wmma_f32_16x16x32_f16(false, va1, false, vb,
                                                  (short)0, acc1, false, false);
  }
  const int ccol = tn * 16 + l;
  const float bv = bias ? bias[ccol] : 0.f;
#pragma unroll
  for (int t = 0; t < 2; ++t) {
    const int base = tm2 * 32 + t * 16;
    const v8f& acc = t ? acc1 : acc0;
    if (base + 16 <= M) {                      // full tile: unconditional
#pragma unroll
      for (int r = 0; r < 8; ++r) {
        const size_t off = (size_t)(base + (half << 3) + r) * Nc + ccol;
        const float v = acc[r] + bv;
        if (Ch) Ch[off] = (_Float16)v; else Cf[off] = v;
      }
    } else {
#pragma unroll
      for (int r = 0; r < 8; ++r) {
        const int crow = base + (half << 3) + r;
        if (crow < M) {
          const size_t off = (size_t)crow * Nc + ccol;
          const float v = acc[r] + bv;
          if (Ch) Ch[off] = (_Float16)v; else Cf[off] = v;
        }
      }
    }
  }
}

// ---------------------------------------------------------------------------
// SR conv (4x4x4 stride 4) + bias + LayerNorm + exact GELU -> xs (f16)
// ---------------------------------------------------------------------------
__global__ __launch_bounds__(128) void sr_conv_ln_gelu_kernel(
    const float* __restrict__ x, const float* __restrict__ sr_w,
    const float* __restrict__ sr_b, const float* __restrict__ g,
    const float* __restrict__ be, _Float16* __restrict__ xs) {
  __shared__ float patch[96 * 64];
  __shared__ float outc[96];
  const int tid = threadIdx.x;
  const int bp  = blockIdx.x;              // 0..685
  const int b   = bp / NK;
  const int p   = bp % NK;
  const int oh = p / 49, ow = (p / 7) % 7, od = p % 7;

  for (int t = tid; t < 96 * 64; t += 128) {
    const int ci = t >> 6;
    const int rem = t & 63;
    const int kh = rem >> 4, kw = (rem >> 2) & 3, kd = rem & 3;
    const int n = ((oh * 4 + kh) * 28 + (ow * 4 + kw)) * 28 + (od * 4 + kd);
    patch[t] = x[((size_t)(b * NN + n)) * CC + ci];
  }
  __syncthreads();

  float acc = 0.f;
  if (tid < 96) {
    const float* wr = sr_w + (size_t)tid * (96 * 64);  // [Cout][Cin*64]
    for (int t = 0; t < 96 * 64; ++t) acc += patch[t] * wr[t];
    acc += sr_b[tid];
    outc[tid] = acc;
  }
  __syncthreads();
  if (tid < 96) {
    float mu = 0.f, sq = 0.f;
    for (int c = 0; c < 96; ++c) { const float v = outc[c]; mu += v; sq += v * v; }
    mu *= (1.f / 96.f);
    const float var = sq * (1.f / 96.f) - mu * mu;
    const float xn = (acc - mu) * rsqrtf(var + 1e-5f);
    const float gg = xn * g[tid] + be[tid];
    const float ge = 0.5f * gg * (1.f + erff(gg * 0.70710678118f));
    xs[(size_t)bp * CC + tid] = (_Float16)ge;
  }
}

// ---------------------------------------------------------------------------
// Depthwise 3x3x3 conv with zero padding: lepe[b,n,c]  (f32 in/out)
// ---------------------------------------------------------------------------
__global__ __launch_bounds__(256) void dwconv_kernel(
    const float* __restrict__ t, const float* __restrict__ cw,
    const float* __restrict__ cb, float* __restrict__ lepe) {
  const int idx = blockIdx.x * 256 + threadIdx.x;
  if (idx >= MR * CC) return;
  const int c  = idx % CC;
  const int bn = idx / CC;
  const int b  = bn / NN;
  const int n  = bn % NN;
  const int hh = n / 784, ww = (n / 28) % 28, dd = n % 28;
  float acc = cb[c];
  const float* wc = cw + c * 27;
  for (int kh = 0; kh < 3; ++kh) {
    const int ih = hh + kh - 1; if (ih < 0 || ih >= 28) continue;
    for (int kw = 0; kw < 3; ++kw) {
      const int iw = ww + kw - 1; if (iw < 0 || iw >= 28) continue;
      for (int kd = 0; kd < 3; ++kd) {
        const int id2 = dd + kd - 1; if (id2 < 0 || id2 >= 28) continue;
        acc += t[((size_t)(b * NN + (ih * 28 + iw) * 28 + id2)) * CC + c] *
               wc[kh * 9 + kw * 3 + kd];
      }
    }
  }
  lepe[idx] = acc;
}

// Shared softmax over one wave's 16x352 score tile (rows split across halves)
__device__ __forceinline__ void wave_softmax(float (*S)[352], int l, int half) {
  float* row = S[l];
  const int c0 = half * 176;
  float mx = -1e30f;
  for (int c = c0; c < c0 + 176; ++c) mx = fmaxf(mx, row[c]);
  mx = fmaxf(mx, __shfl_xor(mx, 16));
  float sum = 0.f;
  for (int c = c0; c < c0 + 176; ++c) {
    const float e = __expf(row[c] - mx);
    row[c] = e;
    sum += e;
  }
  sum += __shfl_xor(sum, 16);
  const float inv = 1.f / sum;
  for (int c = c0; c < c0 + 176; ++c) row[c] *= inv;
}

// Load P tile (A operand) for chunk k0 from LDS scores (f32 -> f16)
__device__ __forceinline__ v16h load_p(float (*S)[352], int l, int half, int k0) {
  const float4* P0 = (const float4*)(&S[l][k0 + (half << 4)]);
  const float4* P1 = (const float4*)(&S[l][k0 + (half << 4) + 8]);
  const float4 p0 = P0[0], p1 = P0[1], p2 = P1[0], p3 = P1[1];
  v16h ap;
  cvt8(p0, p1, ap, 0); cvt8(p2, p3, ap, 8);
  return ap;
}

// Load Q/A operand (f16): 8 valid K values (hd=16 split across halves), rest 0
__device__ __forceinline__ v16h load_q(const _Float16* qp, int half) {
  const v8h f = *(const v8h*)(qp + (half << 3));
  const v8h z = {};
  return CAT16(f, z);
}

// Load K/B operand (f16): 8 valid K values (hd=16 split across halves), rest 0
__device__ __forceinline__ v16h load_k(const _Float16* kp, int half) {
  v16h bk;
#pragma unroll
  for (int i = 0; i < 16; ++i) bk[i] = (_Float16)0.f;
  {
    const v8h f0 = *(const v8h*)(kp + (half << 3));
    const v8h z = {};
    bk = CAT16(f0, z);
  }
  return bk;
}

// ---------------------------------------------------------------------------
// Branch 1: SR global attention.  One wave == 16 query rows for one (b,h).
// ---------------------------------------------------------------------------
__global__ __launch_bounds__(64) void attn_sr_kernel(
    const _Float16* __restrict__ q1, const _Float16* __restrict__ kv1,
    float* __restrict__ y) {
  __shared__ float sc[2][16][352];
  const int w    = threadIdx.x >> 5;
  const int lane = threadIdx.x & 31;
  const int half = lane >> 4;
  const int l    = lane & 15;
  const int id = blockIdx.x * 2 + w;       // 0..8231 (grid exact)
  const int qt = id % QT1;
  const int h  = (id / QT1) % NH2;
  const int b  = id / (QT1 * NH2);
  float (*S)[352] = sc[w];

  const int qrow = qt * 16 + l;
  const v16h aq = load_q(q1 + ((size_t)(b * NN + qrow)) * 48 + h * HD, half);

  // scores S = (Q K^T) * scale
  for (int kt = 0; kt < KT22; ++kt) {
    int key = kt * 16 + l; if (key > NK - 1) key = NK - 1;
    const v16h bk = load_k(kv1 + ((size_t)(b * NK + key)) * CC + h * HD, half);
    v8f s = {};
    s = __builtin_amdgcn_wmma_f32_16x16x32_f16(false, aq, false, bk,
                                               (short)0, s, false, false);
#pragma unroll
    for (int r = 0; r < 8; ++r)
      S[(half << 3) + r][kt * 16 + l] = s[r] * 0.25f;
  }
  __syncthreads();
  for (int idx = lane; idx < 16 * 9; idx += 32)        // mask padded key cols
    S[idx / 9][343 + idx % 9] = -1e30f;
  __syncthreads();
  wave_softmax(S, l, half);
  __syncthreads();

  // O = P V
  v8f o = {};
  for (int kc = 0; kc < 11; ++kc) {
    const int k0 = kc * 32;
    const v16h ap = load_p(S, l, half, k0);
    v16h bv;
#pragma unroll
    for (int i = 0; i < 16; ++i) {
      int key = k0 + (half << 4) + i; if (key > NK - 1) key = NK - 1;  // P col 0 there
      bv[i] = kv1[((size_t)(b * NK + key)) * CC + 48 + h * HD + l];
    }
    o = __builtin_amdgcn_wmma_f32_16x16x32_f16(false, ap, false, bv,
                                               (short)0, o, false, false);
  }
#pragma unroll
  for (int r = 0; r < 8; ++r) {
    const int qr = qt * 16 + (half << 3) + r;
    y[((size_t)(b * NN + qr)) * CC + h * HD + l] = o[r];              // cols 0..47
  }
}

// ---------------------------------------------------------------------------
// Branch 2: 7x7x7 windowed attention. One wave == one 16-query tile of a window.
// ---------------------------------------------------------------------------
__device__ __forceinline__ int win_map(int wx, int wy, int wz, int e) {
  const int i = e / 49;
  const int r = e % 49;
  const int j = r / 7;
  const int k = r % 7;
  return ((wx * 7 + i) * 28 + (wy * 7 + j)) * 28 + (wz * 7 + k);
}

__global__ __launch_bounds__(64) void attn_win_kernel(
    const _Float16* __restrict__ q2, const _Float16* __restrict__ kv2,
    float* __restrict__ y) {
  __shared__ float sc[2][16][352];
  const int w    = threadIdx.x >> 5;
  const int lane = threadIdx.x & 31;
  const int half = lane >> 4;
  const int l    = lane & 15;
  const int id  = blockIdx.x * 2 + w;        // 0..8447 (grid exact)
  const int qt  = id % KT22;
  const int win = (id / KT22) % 64;
  const int h   = (id / (KT22 * 64)) % NH2;
  const int b   = id / (KT22 * 64 * NH2);
  const int wx = win >> 4, wy = (win >> 2) & 3, wz = win & 3;
  float (*S)[352] = sc[w];

  int qe = qt * 16 + l; if (qe > NK - 1) qe = NK - 1;
  const int qn = win_map(wx, wy, wz, qe);
  const v16h aq = load_q(q2 + ((size_t)(b * NN + qn)) * 48 + h * HD, half);

  for (int kt = 0; kt < KT22; ++kt) {
    int ke = kt * 16 + l; if (ke > NK - 1) ke = NK - 1;
    const int kn = win_map(wx, wy, wz, ke);
    const v16h bk = load_k(kv2 + ((size_t)(b * NN + kn)) * CC + h * HD, half);
    v8f s = {};
    s = __builtin_amdgcn_wmma_f32_16x16x32_f16(false, aq, false, bk,
                                               (short)0, s, false, false);
#pragma unroll
    for (int r = 0; r < 8; ++r)
      S[(half << 3) + r][kt * 16 + l] = s[r] * 0.25f;
  }
  __syncthreads();
  for (int idx = lane; idx < 16 * 9; idx += 32)
    S[idx / 9][343 + idx % 9] = -1e30f;
  __syncthreads();
  wave_softmax(S, l, half);
  __syncthreads();

  v8f o = {};
  for (int kc = 0; kc < 11; ++kc) {
    const int k0 = kc * 32;
    const v16h ap = load_p(S, l, half, k0);
    v16h bv;
#pragma unroll
    for (int i = 0; i < 16; ++i) {
      int key = k0 + (half << 4) + i; if (key > NK - 1) key = NK - 1;  // P col 0 there
      const int vn = win_map(wx, wy, wz, key);
      bv[i] = kv2[((size_t)(b * NN + vn)) * CC + 48 + h * HD + l];
    }
    o = __builtin_amdgcn_wmma_f32_16x16x32_f16(false, ap, false, bv,
                                               (short)0, o, false, false);
  }
#pragma unroll
  for (int r = 0; r < 8; ++r) {
    const int qs = qt * 16 + (half << 3) + r;
    if (qs < NK) {
      const int qn2 = win_map(wx, wy, wz, qs);
      y[((size_t)(b * NN + qn2)) * CC + 48 + h * HD + l] = o[r];      // cols 48..95
    }
  }
}

// yh = f16(y + lepe): fused residual add + conversion for the final GEMM
__global__ __launch_bounds__(256) void add_cvt_kernel(
    const float* __restrict__ y, const float* __restrict__ lepe,
    _Float16* __restrict__ yh) {
  const int idx = blockIdx.x * 256 + threadIdx.x;
  if (idx < MR * CC) yh[idx] = (_Float16)(y[idx] + lepe[idx]);
}

// ---------------------------------------------------------------------------
extern "C" void kernel_launch(void* const* d_in, const int* in_sizes, int n_in,
                              void* d_out, int out_size, void* d_ws, size_t ws_size,
                              hipStream_t stream) {
  (void)in_sizes; (void)n_in; (void)out_size; (void)ws_size;
  const float* x           = (const float*)d_in[0];
  const float* lepe_lin_w  = (const float*)d_in[4];
  const float* lepe_lin_b  = (const float*)d_in[5];
  const float* lepe_conv_w = (const float*)d_in[6];
  const float* lepe_conv_b = (const float*)d_in[7];
  const float* sr_w        = (const float*)d_in[8];
  const float* sr_b        = (const float*)d_in[9];
  const float* norm_g      = (const float*)d_in[10];
  const float* norm_b      = (const float*)d_in[11];
  const float* q1_w        = (const float*)d_in[12];
  const float* kv1_w       = (const float*)d_in[13];
  const float* q2_w        = (const float*)d_in[14];
  const float* kv2_w       = (const float*)d_in[15];
  const float* proj_w      = (const float*)d_in[16];
  const float* proj_b      = (const float*)d_in[17];
  float* out = (float*)d_out;

  // ---- workspace layout: f32 region first, then f16 region ----
  float* pf = (float*)d_ws;
  float* lepe_t = pf; pf += (size_t)MR * CC;      // LePE linear output (f32)
  float* lepe   = pf; pf += (size_t)MR * CC;      // LePE conv output   (f32)
  float* yb     = pf; pf += (size_t)MR * CC;      // concat(x1,x2)      (f32)
  _Float16* ph = (_Float16*)pf;
  _Float16* xh      = ph; ph += (size_t)MR * CC;  // x in f16
  _Float16* q1h     = ph; ph += (size_t)MR * 48;
  _Float16* q2h     = ph; ph += (size_t)MR * 48;
  _Float16* kv2h    = ph; ph += (size_t)MR * CC;
  _Float16* yh      = ph; ph += (size_t)MR * CC;  // (y + lepe) in f16
  _Float16* xsh     = ph; ph += (size_t)688 * CC;
  _Float16* kv1h    = ph; ph += (size_t)688 * CC;
  _Float16* lepe_lin_wh = ph; ph += CC * CC;
  _Float16* q1_wh       = ph; ph += 48 * CC;
  _Float16* q2_wh       = ph; ph += 48 * CC;
  _Float16* kv1_wh      = ph; ph += CC * CC;
  _Float16* kv2_wh      = ph; ph += CC * CC;
  _Float16* proj_wh     = ph; ph += CC * CC;

  auto cvt = [&](const float* s, _Float16* d, int n) {
    cvt_kernel<<<(n + 255) / 256, 256, 0, stream>>>(s, d, n);
  };
  auto gemm = [&](const _Float16* A, const _Float16* W, const float* bias,
                  float* Cf, _Float16* Ch, int M, int Nc, int K, int lda) {
    const int mt2 = (((M + 15) >> 4) + 1) >> 1;
    const int waves = mt2 * (Nc >> 4);
    wmma_gemm_f16<<<(waves + 3) >> 2, 128, 0, stream>>>(A, W, bias, Cf, Ch,
                                                        M, Nc, K, lda);
  };

  // One-time f16 conversions (activations + weights)
  cvt(x, xh, MR * CC);
  cvt(lepe_lin_w, lepe_lin_wh, CC * CC);
  cvt(q1_w, q1_wh, 48 * CC);
  cvt(q2_w, q2_wh, 48 * CC);
  cvt(kv1_w, kv1_wh, CC * CC);
  cvt(kv2_w, kv2_wh, CC * CC);
  cvt(proj_w, proj_wh, CC * CC);

  // Projections from x
  gemm(xh, lepe_lin_wh, lepe_lin_b, lepe_t, nullptr, MR, CC, CC, CC);
  gemm(xh, q1_wh,  nullptr, nullptr, q1h,  MR, 48, CC, CC);
  gemm(xh, q2_wh,  nullptr, nullptr, q2h,  MR, 48, CC, CC);
  gemm(xh, kv2_wh, nullptr, nullptr, kv2h, MR, CC, CC, CC);

  // SR conv + LN + GELU (writes f16), then kv1 projection
  sr_conv_ln_gelu_kernel<<<NB * NK, 128, 0, stream>>>(x, sr_w, sr_b,
                                                      norm_g, norm_b, xsh);
  gemm(xsh, kv1_wh, nullptr, nullptr, kv1h, NB * NK, CC, CC, CC);

  // LePE depthwise conv (f32)
  dwconv_kernel<<<(MR * CC + 255) / 256, 256, 0, stream>>>(lepe_t, lepe_conv_w,
                                                           lepe_conv_b, lepe);

  // Attention branches -> fused y buffer (f32)
  attn_sr_kernel<<<(NB * NH2 * QT1) / 2, 64, 0, stream>>>(q1h, kv1h, yb);
  attn_win_kernel<<<(NB * NH2 * 64 * KT22) / 2, 64, 0, stream>>>(q2h, kv2h, yb);

  // yh = f16(y + lepe), then final projection into d_out (f32)
  add_cvt_kernel<<<(MR * CC + 255) / 256, 256, 0, stream>>>(yb, lepe, yh);
  gemm(yh, proj_wh, proj_b, out, nullptr, MR, CC, CC, CC);
}
